// Block_65395172049538
// MI455X (gfx1250) — matrix operand order repaired and checked by reading, not compiled, not run on hardware
//
#include <hip/hip_runtime.h>
#include <hip/hip_bf16.h>

typedef __attribute__((ext_vector_type(16))) __bf16 v16bf;
typedef __attribute__((ext_vector_type(2)))  __bf16 v2bf;
typedef __attribute__((ext_vector_type(8)))  float  v8f;

#define TT 2048
#define CC 2048
#define FF 8192

// Native bf16 convert: lowers to v_cvt_pk_bf16_f32 on gfx1250 (verified in asm).
__device__ __forceinline__ unsigned pack2bf16(float a, float b) {
    v2bf r = { (__bf16)a, (__bf16)b };
    return __builtin_bit_cast(unsigned, r);
}

union Frag16 { v16bf v; unsigned u[8]; };

// Epilogue modes
#define EP_NONE      0
#define EP_SIGMOID   1
#define EP_RELUSQ    2
#define EP_RESID     3
#define EP_RESIDMUL  4

// D[M,N] = epilogue(A[M,K] @ B[N,K]^T)
// A dtype: float or bf16(ushort). B always float. Out: float or bf16(ushort).
// Register-prefetch (float4 / b128 loads) + double-buffered LDS: one barrier per K-step.
template <typename AT, typename OT, int EP>
__global__ __launch_bounds__(256) void gemm_nt_bf16(
    const AT* __restrict__ A, const float* __restrict__ B, OT* __restrict__ D,
    const float* __restrict__ res, const float* __restrict__ mul,
    int M, int N, int K)
{
    constexpr int BM = 128, BN = 128, BK = 32;
    constexpr int LDSS = BK + 8;                    // ushort units; 80 B row stride (8B-aligned)
    __shared__ unsigned short sA[2][BM * LDSS];     // 2 x 10 KB
    __shared__ unsigned short sB[2][BN * LDSS];     // 2 x 10 KB

    const int tid  = threadIdx.x;
    const int lane = tid & 31;
    const int wave = tid >> 5;
    const int wm = wave & 3;          // 4 waves along M (4*32 = 128)
    const int wn = wave >> 2;         // 2 waves along N (2*64 = 128)
    const int h    = lane >> 4;       // lane half
    const int l16  = lane & 15;

    const size_t bM = (size_t)blockIdx.y * BM;
    const size_t bN = (size_t)blockIdx.x * BN;

    v8f acc[2][4];
    const v8f vzero = {0.f,0.f,0.f,0.f,0.f,0.f,0.f,0.f};
    #pragma unroll
    for (int i = 0; i < 2; ++i)
        #pragma unroll
        for (int j = 0; j < 4; ++j) acc[i][j] = vzero;

    // Per-thread prefetch registers: 4 quads per matrix (128x32 tile / 256 threads).
    float4 fA4[4];      // float-A path  (global_load_b128)
    uint2  uA2[4];      // bf16-A path   (global_load_b64)
    float4 fB4[4];

    auto fetch = [&](int k0) {
        #pragma unroll
        for (int i = 0; i < 4; ++i) {
            int q = i * 256 + tid;            // quad index 0..1023
            int row = q >> 3;                 // 8 quads per 32-element row
            int col = (q & 7) * 4;
            size_t ga = (bM + (size_t)row) * (size_t)K + (size_t)(k0 + col);
            if constexpr (sizeof(AT) == 4)
                fA4[i] = *(const float4*)((const float*)A + ga);
            else
                uA2[i] = *(const uint2*)((const unsigned short*)A + ga);
            fB4[i] = *(const float4*)(B + (bN + (size_t)row) * (size_t)K + (size_t)(k0 + col));
        }
    };
    auto commit = [&](int buf) {
        #pragma unroll
        for (int i = 0; i < 4; ++i) {
            int q = i * 256 + tid, row = q >> 3, col = (q & 7) * 4;
            uint2 pa;
            if constexpr (sizeof(AT) == 4)
                pa = make_uint2(pack2bf16(fA4[i].x, fA4[i].y), pack2bf16(fA4[i].z, fA4[i].w));
            else
                pa = uA2[i];
            *(uint2*)&sA[buf][row * LDSS + col] = pa;
            uint2 pb = make_uint2(pack2bf16(fB4[i].x, fB4[i].y), pack2bf16(fB4[i].z, fB4[i].w));
            *(uint2*)&sB[buf][row * LDSS + col] = pb;
        }
    };

    const int NK = K / BK;
    fetch(0);
    commit(0);
    __syncthreads();

    for (int kt = 0; kt < NK; ++kt) {
        const int buf = kt & 1;
        if (kt + 1 < NK) fetch((kt + 1) * BK);   // overlaps with WMMAs below

        // ---- fragments from current LDS buffer ----
        Frag16 af[2], bfr[4];
        #pragma unroll
        for (int mt = 0; mt < 2; ++mt) {
            int m = wm * 32 + mt * 16 + l16;     // A lane row (M)
            #pragma unroll
            for (int p = 0; p < 8; ++p) {
                int kk = ((p < 4) ? 2 * p : 16 + 2 * (p - 4)) + 8 * h;  // ISA 16-bit A layout
                af[mt].u[p] = *(const unsigned*)&sA[buf][m * LDSS + kk];
            }
        }
        #pragma unroll
        for (int nt = 0; nt < 4; ++nt) {
            int n = wn * 64 + nt * 16 + l16;     // B lane row (N)
            #pragma unroll
            for (int p = 0; p < 8; ++p) {
                int kk = ((p < 4) ? 2 * p : 16 + 2 * (p - 4)) + 8 * h;  // ISA 16-bit B layout (same K order as A)
                bfr[nt].u[p] = *(const unsigned*)&sB[buf][n * LDSS + kk];
            }
        }

        #pragma unroll
        for (int mt = 0; mt < 2; ++mt)
            #pragma unroll
            for (int nt = 0; nt < 4; ++nt)
                acc[mt][nt] = __builtin_amdgcn_wmma_f32_16x16x32_bf16(
                    false, af[mt].v, false, bfr[nt].v,
                    (short)0, acc[mt][nt], false, false);

        if (kt + 1 < NK) commit(buf ^ 1);        // write only the idle buffer
        __syncthreads();
    }

    // ---- epilogue + store: D element at (M = r + 8*h, N = l16) per C/D layout ----
    #pragma unroll
    for (int mt = 0; mt < 2; ++mt) {
        #pragma unroll
        for (int nt = 0; nt < 4; ++nt) {
            #pragma unroll
            for (int r = 0; r < 8; ++r) {
                size_t row = bM + (size_t)(wm * 32 + mt * 16 + r + 8 * h);
                size_t col = bN + (size_t)(wn * 64 + nt * 16 + l16);
                size_t idx = row * (size_t)N + col;
                float v = acc[mt][nt][r];
                if constexpr (EP == EP_SIGMOID)  v = 1.f / (1.f + __expf(-v));
                if constexpr (EP == EP_RELUSQ)   { float t = fmaxf(v, 0.f); v = t * t; }
                if constexpr (EP == EP_RESID)    v = res[idx] + v;
                if constexpr (EP == EP_RESIDMUL) v = res[idx] + mul[idx] * v;
                if constexpr (sizeof(OT) == 4) ((float*)D)[idx] = v;
                else ((__bf16*)D)[idx] = (__bf16)v;
            }
        }
    }
}

// ---- LayerNorm over rows: out[t,:] = (x - mean)*rsqrt(var+1e-5)*w + b ----
__global__ __launch_bounds__(256) void ln_rows(
    const float* __restrict__ x, const float* __restrict__ w,
    const float* __restrict__ b, float* __restrict__ out)
{
    const int t = blockIdx.x;
    const float* row = x + (size_t)t * CC;
    float s = 0.f, s2 = 0.f;
    for (int c = threadIdx.x; c < CC; c += 256) { float v = row[c]; s += v; s2 += v * v; }
    __shared__ float sh[256], sh2[256];
    sh[threadIdx.x] = s; sh2[threadIdx.x] = s2;
    __syncthreads();
    for (int off = 128; off > 0; off >>= 1) {
        if (threadIdx.x < off) { sh[threadIdx.x] += sh[threadIdx.x + off];
                                 sh2[threadIdx.x] += sh2[threadIdx.x + off]; }
        __syncthreads();
    }
    const float mean = sh[0] * (1.f / CC);
    const float var  = sh2[0] * (1.f / CC) - mean * mean;
    const float inv  = rsqrtf(var + 1e-5f);
    float* orow = out + (size_t)t * CC;
    for (int c = threadIdx.x; c < CC; c += 256)
        orow[c] = (row[c] - mean) * inv * w[c] + b[c];
}

// ---- token-shift mix, 3 outputs (att) ----
__global__ __launch_bounds__(256) void mix3(
    const float* __restrict__ rx, const float* __restrict__ shift,
    const float* __restrict__ tmk, const float* __restrict__ tmv, const float* __restrict__ tmr,
    float* __restrict__ xk, float* __restrict__ xv, float* __restrict__ xr)
{
    size_t i = (size_t)blockIdx.x * 256 + threadIdx.x;
    int t = (int)(i / CC), c = (int)(i % CC);
    float cur  = rx[i];
    float prev = t ? rx[i - CC] : shift[c];
    float mk = tmk[c], mv = tmv[c], mr = tmr[c];
    xk[i] = cur * mk + prev * (1.f - mk);
    xv[i] = cur * mv + prev * (1.f - mv);
    xr[i] = cur * mr + prev * (1.f - mr);
}

// ---- token-shift mix, 2 outputs (ffn) ----
__global__ __launch_bounds__(256) void mix2(
    const float* __restrict__ fx, const float* __restrict__ shift,
    const float* __restrict__ tmk, const float* __restrict__ tmr,
    float* __restrict__ fk, float* __restrict__ fr)
{
    size_t i = (size_t)blockIdx.x * 256 + threadIdx.x;
    int t = (int)(i / CC), c = (int)(i % CC);
    float cur  = fx[i];
    float prev = t ? fx[i - CC] : shift[c];
    float mk = tmk[c], mr = tmr[c];
    fk[i] = cur * mk + prev * (1.f - mk);
    fr[i] = cur * mr + prev * (1.f - mr);
}

// ---- WKV serial scan: per-channel recurrence, f64 math + f32 state round-trip
//      (faithful to reference). Writes sry = sr * y directly. ----
__global__ __launch_bounds__(256) void wkv_scan(
    const float* __restrict__ k, const float* __restrict__ v,
    const float* __restrict__ time_decay, const float* __restrict__ time_first,
    const float* __restrict__ wkv_state, const float* __restrict__ sr,
    float* __restrict__ sry)
{
    const int c = blockIdx.x * 256 + threadIdx.x;
    const double u  = (double)time_first[c];
    const double w  = -exp((double)time_decay[c]);
    const double ew = exp(w);
    float a = wkv_state[c];          // state[0]
    float b = wkv_state[CC + c];     // state[1]
    for (int t = 0; t < TT; ++t) {
        const size_t i = (size_t)t * CC + c;
        const double kd = (double)k[i];
        const double vd = (double)v[i];
        const double e  = exp(u + w + kd);
        const double y  = ((double)a + e * vd) / ((double)b + e);
        const double kk = exp(kd);
        a = (float)(((double)a + kk * vd) * ew);
        b = (float)(((double)b + kk) * ew);
        sry[i] = sr[i] * (float)y;
    }
}

extern "C" void kernel_launch(void* const* d_in, const int* in_sizes, int n_in,
                              void* d_out, int out_size, void* d_ws, size_t ws_size,
                              hipStream_t stream) {
    const float* x       = (const float*)d_in[0];
    const float* attsh   = (const float*)d_in[1];
    const float* wkvst   = (const float*)d_in[2];
    const float* ffnsh   = (const float*)d_in[3];
    const float* ln1w    = (const float*)d_in[4];
    const float* ln1b    = (const float*)d_in[5];
    const float* ln2w    = (const float*)d_in[6];
    const float* ln2b    = (const float*)d_in[7];
    const float* tmk     = (const float*)d_in[8];
    const float* tmv     = (const float*)d_in[9];
    const float* tmr     = (const float*)d_in[10];
    const float* tdecay  = (const float*)d_in[11];
    const float* tfirst  = (const float*)d_in[12];
    const float* Wk      = (const float*)d_in[13];
    const float* Wv      = (const float*)d_in[14];
    const float* Wr      = (const float*)d_in[15];
    const float* Wo      = (const float*)d_in[16];
    const float* ftmk    = (const float*)d_in[17];
    const float* ftmr    = (const float*)d_in[18];
    const float* Wfk     = (const float*)d_in[19];  // [F,C]
    const float* Wfv     = (const float*)d_in[20];  // [C,F]
    const float* Wfr     = (const float*)d_in[21];
    float* out = (float*)d_out;

    const size_t TCB = (size_t)TT * CC * sizeof(float);    // 16 MB
    char* ws = (char*)d_ws;
    float* p_rx  = (float*)(ws + 0 * TCB);   // rx  -> later sr
    float* p_xk  = (float*)(ws + 1 * TCB);   // xk  -> later x1
    float* p_xv  = (float*)(ws + 2 * TCB);   // xv  -> later fk
    float* p_xr  = (float*)(ws + 3 * TCB);   // xr  -> later fr
    float* p_k   = (float*)(ws + 4 * TCB);   // k   -> later r2
    float* p_v   = (float*)(ws + 5 * TCB);   // v   -> later fx
    float* p_sry = (float*)(ws + 6 * TCB);   // sr*y
    unsigned short* p_kf = (unsigned short*)(ws + 7 * TCB); // kf bf16 [T,F], 32 MB

    const dim3 blk(256);
    const dim3 gEW((TT * (size_t)CC) / 256);
    const dim3 gCC(CC / 128, TT / 128);
    const dim3 gCF(FF / 128, TT / 128);

    // --- attention half ---
    ln_rows<<<TT, blk, 0, stream>>>(x, ln1w, ln1b, p_rx);
    mix3<<<gEW, blk, 0, stream>>>(p_rx, attsh, tmk, tmv, tmr, p_xk, p_xv, p_xr);
    gemm_nt_bf16<float, float, EP_NONE><<<gCC, blk, 0, stream>>>(
        p_xk, Wk, p_k, nullptr, nullptr, TT, CC, CC);
    gemm_nt_bf16<float, float, EP_NONE><<<gCC, blk, 0, stream>>>(
        p_xv, Wv, p_v, nullptr, nullptr, TT, CC, CC);
    gemm_nt_bf16<float, float, EP_SIGMOID><<<gCC, blk, 0, stream>>>(
        p_xr, Wr, p_rx, nullptr, nullptr, TT, CC, CC);          // sr -> p_rx
    wkv_scan<<<CC / 256, blk, 0, stream>>>(p_k, p_v, tdecay, tfirst, wkvst, p_rx, p_sry);
    gemm_nt_bf16<float, float, EP_RESID><<<gCC, blk, 0, stream>>>(
        p_sry, Wo, p_xk, x, nullptr, TT, CC, CC);               // x1 -> p_xk

    // --- channel-mix half ---
    ln_rows<<<TT, blk, 0, stream>>>(p_xk, ln2w, ln2b, p_v);     // fx -> p_v
    mix2<<<gEW, blk, 0, stream>>>(p_v, ffnsh, ftmk, ftmr, p_xv, p_xr); // fk, fr
    gemm_nt_bf16<float, unsigned short, EP_RELUSQ><<<gCF, blk, 0, stream>>>(
        p_xv, Wfk, p_kf, nullptr, nullptr, TT, FF, CC);         // kf bf16 [T,F]
    gemm_nt_bf16<float, float, EP_SIGMOID><<<gCC, blk, 0, stream>>>(
        p_xr, Wfr, p_k, nullptr, nullptr, TT, CC, CC);          // r2 -> p_k
    gemm_nt_bf16<unsigned short, float, EP_RESIDMUL><<<gCC, blk, 0, stream>>>(
        p_kf, Wfv, out, p_xk, p_k, TT, CC, FF);                 // out = x1 + r2*kv
}
